// RnaMsmPkmLayer_4277787427283
// MI455X (gfx1250) — matrix operand order repaired and checked by reading, not compiled, not run on hardware
//
#include <hip/hip_runtime.h>
#include <hip/hip_bf16.h>
#include <math.h>
#include <stdint.h>

// ---- problem constants (match reference) ----
#define BB   2
#define TT   1024
#define DD   768
#define HH   12
#define DHA  64
#define PHN  4
#define NKN  256
#define TOPK 32
#define DHK  128
#define DQW  1024
#define NTOK (BB*TT)
#define EPSL 1e-5f

typedef __attribute__((ext_vector_type(16))) __bf16    v16bf;
typedef __attribute__((ext_vector_type(8)))  float     v8f;
typedef __attribute__((ext_vector_type(2)))  float     v2f;
typedef __attribute__((ext_vector_type(4)))  unsigned  v4u;
typedef __attribute__((ext_vector_type(8)))  unsigned  v8u;

// ---------------- bf16 helpers (bit-level) ---------------------------------
__device__ inline __bf16 to_bf16(float f) {
  unsigned u = __float_as_uint(f);
  u += 0x7FFFu + ((u >> 16) & 1u);            // round-to-nearest-even
  unsigned short h = (unsigned short)(u >> 16);
  return __builtin_bit_cast(__bf16, h);
}

// ---------------- WMMA bf16 fragment loaders (contiguous order: element s = k 16*half + s) ----------
// A 16x32 bf16, row-major, 16B-aligned rows: two contiguous 16B chunks/lane.
__device__ inline v16bf load_a_frag(const __bf16* A, int lda, int row0, int k0, int lane) {
  int half = lane >> 4;
  int m = row0 + (lane & 15);
  const v4u* p = (const v4u*)(A + (size_t)m * lda + k0);
  v4u lo = p[2 * half];        // K = half*16 .. half*16+7
  v4u hi = p[2 * half + 1];    // K = half*16+8 .. half*16+15
  v8u packed;
  packed[0] = lo[0]; packed[1] = lo[1]; packed[2] = lo[2]; packed[3] = lo[3];
  packed[4] = hi[0]; packed[5] = hi[1]; packed[6] = hi[2]; packed[7] = hi[3];
  return __builtin_bit_cast(v16bf, packed);
}
// B 32x16 where logical B[k][n] = M[n][k] (transposed source, row-major M,
// 32B-aligned rows): one contiguous 32-byte read per lane.
__device__ inline v16bf load_bT_frag(const __bf16* M, int ldm, int k0, int col0, int lane) {
  int half = lane >> 4;
  int n = col0 + (lane & 15);
  return *(const v16bf*)(M + (size_t)n * ldm + k0 + half * 16);
}

// ---------------- WMMA f32 16x16x4 fragment loaders ------------------------
__device__ inline v2f load_a4_frag(const float* A, int lda, int row0, int k0, int lane) {
  int half = lane >> 4;
  const float* p = A + (size_t)(row0 + (lane & 15)) * lda + k0 + half * 2;
  v2f a; a[0] = p[0]; a[1] = p[1]; return a;
}
__device__ inline v2f load_b4T_frag(const float* M, int ldm, int k0, int col0, int lane) {
  int half = lane >> 4;
  const float* p = M + (size_t)(col0 + (lane & 15)) * ldm + k0 + half * 2;
  v2f b; b[0] = p[0]; b[1] = p[1]; return b;
}

// ---------------- fp32 -> bf16 transposed convert (weights) ----------------
__global__ void cvt_bf16_T(const float* __restrict__ in, __bf16* __restrict__ out,
                           int Kdim, int Ndim) {
  int i = blockIdx.x * 256 + threadIdx.x;
  if (i < Kdim * Ndim) {
    int k = i / Ndim, n = i % Ndim;
    out[(size_t)n * Kdim + k] = to_bf16(in[i]);
  }
}

// ---------------- LayerNorm (row per block), bf16 out ----------------------
__global__ __launch_bounds__(256) void ln_kernel(const float* __restrict__ x,
                                                 const float* __restrict__ g,
                                                 const float* __restrict__ bta,
                                                 __bf16* __restrict__ out) {
  int row = blockIdx.x, tid = threadIdx.x;
  __shared__ float red[256];
  const float* xr = x + (size_t)row * DD;
  float v0 = xr[tid], v1 = xr[tid + 256], v2 = xr[tid + 512];
  red[tid] = v0 + v1 + v2; __syncthreads();
  for (int off = 128; off; off >>= 1) { if (tid < off) red[tid] += red[tid + off]; __syncthreads(); }
  float mean = red[0] / DD;
  __syncthreads();
  float d0 = v0 - mean, d1 = v1 - mean, d2 = v2 - mean;
  red[tid] = d0 * d0 + d1 * d1 + d2 * d2; __syncthreads();
  for (int off = 128; off; off >>= 1) { if (tid < off) red[tid] += red[tid + off]; __syncthreads(); }
  float rstd = rsqrtf(red[0] / DD + EPSL);
  __bf16* orow = out + (size_t)row * DD;
  orow[tid]       = to_bf16(d0 * rstd * g[tid]       + bta[tid]);
  orow[tid + 256] = to_bf16(d1 * rstd * g[tid + 256] + bta[tid + 256]);
  orow[tid + 512] = to_bf16(d2 * rstd * g[tid + 512] + bta[tid + 512]);
}

// ---------------- WMMA GEMM: 4-wave block computes 64x64 -------------------
// B panel staged in LDS via GLOBAL_LOAD_ASYNC_TO_LDS_B128 and shared by all
// 4 waves (4x less L2 traffic on the weight matrix). Bt is [N x K] row-major.
template <int K>
__global__ __launch_bounds__(128) void gemm_bf16(const __bf16* __restrict__ A,
                                                 const __bf16* __restrict__ Bt,
                                                 const float* __restrict__ bias,
                                                 const float* __restrict__ resid,
                                                 float* __restrict__ Cf,
                                                 __bf16* __restrict__ Cb,
                                                 __bf16* __restrict__ CbT,
                                                 int M, int N) {
  __shared__ __bf16 bt_tile[64 * 32];                 // 4KB: [n_local][k_local]
  int tid = threadIdx.x;
  int wave = tid >> 5, lane = tid & 31;
  int col0 = blockIdx.x * 64;
  int row0 = blockIdx.y * 64 + wave * 16;
  v8f c0 = {}, c1 = {}, c2 = {}, c3 = {};
  for (int k0 = 0; k0 < K; k0 += 32) {
    __syncthreads();                                  // tile consumed by all waves
    // async-stage Bt[col0..col0+63][k0..k0+31] -> LDS (2 x 16B per thread)
#pragma unroll
    for (int i = 0; i < 2; ++i) {
      int cchunk = tid + 128 * i;                     // 256 chunks of 16B
      int nl = cchunk >> 2, kc = (cchunk & 3) * 8;
      unsigned lds_addr = (unsigned)(uintptr_t)&bt_tile[nl * 32 + kc];
      unsigned long long gaddr =
          (unsigned long long)(uintptr_t)(Bt + (size_t)(col0 + nl) * K + k0 + kc);
      asm volatile("global_load_async_to_lds_b128 %0, %1, off"
                   :: "v"(lds_addr), "v"(gaddr) : "memory");
    }
    asm volatile("s_wait_asynccnt 0x0" ::: "memory"); // my async ops landed
    __syncthreads();                                  // everyone's landed
    v16bf a  = load_a_frag(A, K, row0, k0, lane);
    v16bf b0 = load_bT_frag(bt_tile, 32, 0, 0,  lane);
    v16bf b1 = load_bT_frag(bt_tile, 32, 0, 16, lane);
    v16bf b2 = load_bT_frag(bt_tile, 32, 0, 32, lane);
    v16bf b3 = load_bT_frag(bt_tile, 32, 0, 48, lane);
    c0 = __builtin_amdgcn_wmma_f32_16x16x32_bf16(false, a, false, b0, (short)0, c0, false, false);
    c1 = __builtin_amdgcn_wmma_f32_16x16x32_bf16(false, a, false, b1, (short)0, c1, false, false);
    c2 = __builtin_amdgcn_wmma_f32_16x16x32_bf16(false, a, false, b2, (short)0, c2, false, false);
    c3 = __builtin_amdgcn_wmma_f32_16x16x32_bf16(false, a, false, b3, (short)0, c3, false, false);
  }
  int half = lane >> 4;
  v8f cc[4] = {c0, c1, c2, c3};
#pragma unroll
  for (int t = 0; t < 4; ++t) {
    int n = col0 + t * 16 + (lane & 15);
    float bval = bias ? bias[n] : 0.f;
#pragma unroll
    for (int r = 0; r < 8; ++r) {
      int m = row0 + r + 8 * half;
      float val = cc[t][r] + bval;
      if (resid) val += resid[(size_t)m * N + n];
      if (Cf)  Cf[(size_t)m * N + n] = val;
      if (Cb)  Cb[(size_t)m * N + n] = to_bf16(val);
      if (CbT) CbT[(size_t)n * M + m] = to_bf16(val);
    }
  }
}

// ---------------- attention: wave per (b, h, 16-query tile) ----------------
__global__ __launch_bounds__(32) void attn_kernel(const __bf16* __restrict__ Q,
                                                  const __bf16* __restrict__ Km,
                                                  const __bf16* __restrict__ Vt,
                                                  __bf16* __restrict__ O) {
  __shared__ float sc[16 * TT];                       // 64KB scores, reused for bf16 probs
  int lane = threadIdx.x;
  int qt = blockIdx.x, h = blockIdx.y, bb = blockIdx.z;
  int q0 = qt * 16;
  const __bf16* Qb  = Q  + ((size_t)bb * TT) * DD + h * DHA;
  const __bf16* Kb  = Km + ((size_t)bb * TT) * DD + h * DHA;
  const __bf16* Vtb = Vt + (size_t)(h * DHA) * NTOK + (size_t)bb * TT;
  int half = lane >> 4, nc = lane & 15;

  // scores = (Q @ K^T) * 1/sqrt(64)
#pragma unroll 2
  for (int kt = 0; kt < TT / 16; ++kt) {
    v8f c = {};
#pragma unroll
    for (int ks = 0; ks < DHA / 32; ++ks) {
      v16bf a = load_a_frag(Qb, DD, q0, ks * 32, lane);
      v16bf b = load_bT_frag(Kb, DD, ks * 32, kt * 16, lane);
      c = __builtin_amdgcn_wmma_f32_16x16x32_bf16(false, a, false, b, (short)0, c, false, false);
    }
#pragma unroll
    for (int r = 0; r < 8; ++r)
      sc[(r + 8 * half) * TT + kt * 16 + nc] = c[r] * 0.125f;
  }
  __syncthreads();

  // per-row softmax; convert in place to bf16 probs
  __bf16* pb = reinterpret_cast<__bf16*>(sc);
  for (int r = 0; r < 16; ++r) {
    float v[32];
#pragma unroll
    for (int j = 0; j < 32; ++j) v[j] = sc[r * TT + lane + 32 * j];
    float m = -3.4e38f;
#pragma unroll
    for (int j = 0; j < 32; ++j) m = fmaxf(m, v[j]);
    for (int off = 16; off; off >>= 1) m = fmaxf(m, __shfl_xor(m, off));
    float s = 0.f;
#pragma unroll
    for (int j = 0; j < 32; ++j) { v[j] = expf(v[j] - m); s += v[j]; }
    for (int off = 16; off; off >>= 1) s += __shfl_xor(s, off);
    float inv = 1.f / s;
    __syncthreads();                                  // order float loads vs bf16 stores
#pragma unroll
    for (int j = 0; j < 32; ++j) pb[r * TT + lane + 32 * j] = to_bf16(v[j] * inv);
    __syncthreads();
  }

  // out = P @ V  (B fragments from transposed V: contiguous 32B reads)
#pragma unroll
  for (int nt = 0; nt < DHA / 16; ++nt) {
    v8f c = {};
#pragma unroll 2
    for (int ks = 0; ks < TT / 32; ++ks) {
      v16bf a = load_a_frag(pb, TT, 0, ks * 32, lane);
      v16bf b = load_bT_frag(Vtb, NTOK, ks * 32, nt * 16, lane);
      c = __builtin_amdgcn_wmma_f32_16x16x32_bf16(false, a, false, b, (short)0, c, false, false);
    }
#pragma unroll
    for (int r = 0; r < 8; ++r) {
      int m = q0 + r + 8 * half;
      O[((size_t)bb * TT + m) * DD + h * DHA + nt * 16 + nc] = to_bf16(c[r]);
    }
  }
}

// ---------------- BatchNorm over tokens ------------------------------------
__global__ __launch_bounds__(256) void bn_stats(const float* __restrict__ qf,
                                                float* __restrict__ mu,
                                                float* __restrict__ rs) {
  int c = blockIdx.x, tid = threadIdx.x;
  __shared__ float s1[256], s2[256];
  float a = 0.f, b = 0.f;
  for (int t = tid; t < NTOK; t += 256) {
    float v = qf[(size_t)t * DQW + c];
    a += v; b += v * v;
  }
  s1[tid] = a; s2[tid] = b; __syncthreads();
  for (int off = 128; off; off >>= 1) {
    if (tid < off) { s1[tid] += s1[tid + off]; s2[tid] += s2[tid + off]; }
    __syncthreads();
  }
  if (tid == 0) {
    float m = s1[0] / NTOK;
    float v = s2[0] / NTOK - m * m;
    mu[c] = m; rs[c] = rsqrtf(v + EPSL);
  }
}
__global__ void bn_apply(const float* __restrict__ qf, const float* __restrict__ mu,
                         const float* __restrict__ rs, const float* __restrict__ g,
                         const float* __restrict__ bta, float* __restrict__ qn) {
  size_t i = (size_t)blockIdx.x * 256 + threadIdx.x;
  if (i >= (size_t)NTOK * DQW) return;
  int c = (int)(i % DQW);
  qn[i] = (qf[i] - mu[c]) * rs[c] * g[c] + bta[c];
}

// ---------------- PKM sub-key dots on the fp32 matrix pipe -----------------
__global__ __launch_bounds__(32) void pkm_dots_wmma(const float* __restrict__ qn,
                                                    const float* __restrict__ keys,
                                                    float* __restrict__ dots) {
  int lane = threadIdx.x;
  int ntile = blockIdx.x, ktile = blockIdx.y;
  int ph = blockIdx.z >> 1, sub = blockIdx.z & 1;
  const float* Aq = qn + sub * (PHN * DHK) + ph * DHK;                 // lda = DQW
  const float* Kb = keys + ((size_t)ph * NKN * 2 + sub) * DHK;         // ldm = 2*DHK
  int n0 = ntile * 16, key0 = ktile * 16;
  v8f c = {};
#pragma unroll 8
  for (int k0 = 0; k0 < DHK; k0 += 4) {
    v2f a = load_a4_frag(Aq, DQW, n0, k0, lane);
    v2f b = load_b4T_frag(Kb, 2 * DHK, k0, key0, lane);
    c = __builtin_amdgcn_wmma_f32_16x16x4_f32(false, a, false, b, (short)0, c, false, false);
  }
  int half = lane >> 4, nc = lane & 15;
#pragma unroll
  for (int r = 0; r < 8; ++r) {
    int m = n0 + r + 8 * half;
    dots[(((size_t)m * PHN + ph) * 2 + sub) * NKN + key0 + nc] = c[r];
  }
}

// ---------------- top-32 of 256 per (n,ph,sub) -----------------------------
__global__ __launch_bounds__(32) void pkm_topk(const float* __restrict__ dots,
                                               float* __restrict__ s_out,
                                               int* __restrict__ i_out) {
  int n = blockIdx.x, ph = blockIdx.y, sub = blockIdx.z;
  int lane = threadIdx.x;
  const float* dp = dots + (((size_t)n * PHN + ph) * 2 + sub) * NKN;
  float s[8];                                          // key = j*32 + lane
#pragma unroll
  for (int j = 0; j < 8; ++j) s[j] = dp[j * 32 + lane];
  float* so = s_out + (((size_t)n * PHN + ph) * 2 + sub) * TOPK;
  int*   io = i_out + (((size_t)n * PHN + ph) * 2 + sub) * TOPK;
  for (int t = 0; t < TOPK; ++t) {
    float bv = -3.4e38f; int bj = 0;
#pragma unroll
    for (int j = 0; j < 8; ++j) if (s[j] > bv) { bv = s[j]; bj = j; }
    int bidx = bj * 32 + lane;
    for (int off = 16; off; off >>= 1) {
      float ov = __shfl_xor(bv, off); int oi = __shfl_xor(bidx, off);
      if (ov > bv || (ov == bv && oi < bidx)) { bv = ov; bidx = oi; }
    }
    if (lane == 0) { so[t] = bv; io[t] = bidx; }
    if ((bidx & 31) == lane) s[bidx >> 5] = -3.4e38f;
  }
}

// ---------------- PKM combine: cartesian top-32, softmax, gather+resid -----
__global__ __launch_bounds__(256) void pkm_combine(const float* __restrict__ s_in,
                                                   const int* __restrict__ i_in,
                                                   const float* __restrict__ values,
                                                   const float* __restrict__ x1,
                                                   float* __restrict__ out) {
  int n = blockIdx.x, tid = threadIdx.x;
  int wave = tid >> 5, lane = tid & 31;
  __shared__ float s0[PHN][TOPK], s1[PHN][TOPK], attn[PHN][TOPK];
  __shared__ int   i0[PHN][TOPK], i1[PHN][TOPK], vidx[PHN][TOPK];
  {
    int ph = tid >> 6, sub = (tid >> 5) & 1, k = tid & 31;
    size_t off = (((size_t)n * PHN + ph) * 2 + sub) * TOPK + k;
    if (sub == 0) { s0[ph][k] = s_in[off]; i0[ph][k] = i_in[off]; }
    else          { s1[ph][k] = s_in[off]; i1[ph][k] = i_in[off]; }
  }
  __syncthreads();
  if (wave < PHN) {
    int ph = wave;
    float s0v = s0[ph][lane];
    unsigned used = 0;
    for (int t = 0; t < TOPK; ++t) {
      float bv = -3.4e38f; int bj = 0;
#pragma unroll
      for (int j = 0; j < 32; ++j)
        if (!((used >> j) & 1u)) { float v = s0v + s1[ph][j]; if (v > bv) { bv = v; bj = j; } }
      int bidx = lane * 32 + bj;
      for (int off = 16; off; off >>= 1) {
        float ov = __shfl_xor(bv, off); int oi = __shfl_xor(bidx, off);
        if (ov > bv || (ov == bv && oi < bidx)) { bv = ov; bidx = oi; }
      }
      if (lane == 0) {
        attn[ph][t] = bv;
        vidx[ph][t] = i0[ph][bidx >> 5] * NKN + i1[ph][bidx & 31];
      }
      if ((bidx >> 5) == lane) used |= 1u << (bidx & 31);
    }
  }
  __syncthreads();
  if (wave < PHN) {
    int ph = wave;
    float v = attn[ph][lane];
    float m = v;
    for (int off = 16; off; off >>= 1) m = fmaxf(m, __shfl_xor(m, off));
    float e = expf(v - m);
    float s = e;
    for (int off = 16; off; off >>= 1) s += __shfl_xor(s, off);
    attn[ph][lane] = e / s;
  }
  __syncthreads();
  if (tid < PHN * TOPK) {                              // global_prefetch_b8 path
    int ph = tid >> 5, k = tid & 31;
    __builtin_prefetch(values + (size_t)vidx[ph][k] * DD, 0, 3);
  }
  for (int d = tid; d < DD; d += 256) {
    float acc = x1[(size_t)n * DD + d];
    for (int ph = 0; ph < PHN; ++ph)
#pragma unroll 4
      for (int k = 0; k < TOPK; ++k)
        acc += attn[ph][k] * values[(size_t)vidx[ph][k] * DD + d];
    out[(size_t)n * DD + d] = acc;
  }
}

// ---------------- host orchestration ---------------------------------------
extern "C" void kernel_launch(void* const* d_in, const int* in_sizes, int n_in,
                              void* d_out, int out_size, void* d_ws, size_t ws_size,
                              hipStream_t stream) {
  (void)in_sizes; (void)n_in; (void)out_size; (void)ws_size;
  const float* x      = (const float*)d_in[0];
  const float* wq     = (const float*)d_in[1];
  const float* bq     = (const float*)d_in[2];
  const float* wk     = (const float*)d_in[3];
  const float* bk     = (const float*)d_in[4];
  const float* wv     = (const float*)d_in[5];
  const float* bv     = (const float*)d_in[6];
  const float* wo     = (const float*)d_in[7];
  const float* bo     = (const float*)d_in[8];
  const float* ln1_g  = (const float*)d_in[9];
  const float* ln1_b  = (const float*)d_in[10];
  const float* ln2_g  = (const float*)d_in[11];
  const float* ln2_b  = (const float*)d_in[12];
  const float* pkm_wq = (const float*)d_in[13];
  const float* bn_g   = (const float*)d_in[14];
  const float* bn_b   = (const float*)d_in[15];
  const float* keys   = (const float*)d_in[16];
  const float* values = (const float*)d_in[17];

  char* ws = (char*)d_ws;
  auto alloc = [&](size_t bytes) {
    char* p = ws; ws += (bytes + 255) & ~(size_t)255; return p;
  };
  __bf16* wq_bf  = (__bf16*)alloc((size_t)DD * DD * 2);     // transposed
  __bf16* wk_bf  = (__bf16*)alloc((size_t)DD * DD * 2);
  __bf16* wv_bf  = (__bf16*)alloc((size_t)DD * DD * 2);
  __bf16* wo_bf  = (__bf16*)alloc((size_t)DD * DD * 2);
  __bf16* pwq_bf = (__bf16*)alloc((size_t)DD * DQW * 2);    // transposed [DQW x DD]
  __bf16* h_bf   = (__bf16*)alloc((size_t)NTOK * DD * 2);
  __bf16* q_bf   = (__bf16*)alloc((size_t)NTOK * DD * 2);
  __bf16* k_bf   = (__bf16*)alloc((size_t)NTOK * DD * 2);
  __bf16* vt_bf  = (__bf16*)alloc((size_t)NTOK * DD * 2);   // transposed [DD x NTOK]
  __bf16* ao_bf  = (__bf16*)alloc((size_t)NTOK * DD * 2);
  float*  x1     = (float*)alloc((size_t)NTOK * DD * 4);
  __bf16* h2_bf  = (__bf16*)alloc((size_t)NTOK * DD * 2);
  float*  qf     = (float*)alloc((size_t)NTOK * DQW * 4);
  float*  qn     = (float*)alloc((size_t)NTOK * DQW * 4);
  float*  mu     = (float*)alloc((size_t)DQW * 4);
  float*  rs     = (float*)alloc((size_t)DQW * 4);
  float*  dots   = (float*)alloc((size_t)NTOK * PHN * 2 * NKN * 4);
  float*  ts     = (float*)alloc((size_t)NTOK * PHN * 2 * TOPK * 4);
  int*    ti     = (int*)alloc((size_t)NTOK * PHN * 2 * TOPK * 4);

  // 1) weights -> bf16, transposed
  cvt_bf16_T<<<(DD * DD + 255) / 256, 256, 0, stream>>>(wq, wq_bf, DD, DD);
  cvt_bf16_T<<<(DD * DD + 255) / 256, 256, 0, stream>>>(wk, wk_bf, DD, DD);
  cvt_bf16_T<<<(DD * DD + 255) / 256, 256, 0, stream>>>(wv, wv_bf, DD, DD);
  cvt_bf16_T<<<(DD * DD + 255) / 256, 256, 0, stream>>>(wo, wo_bf, DD, DD);
  cvt_bf16_T<<<(DD * DQW + 255) / 256, 256, 0, stream>>>(pkm_wq, pwq_bf, DD, DQW);

  // 2) LN1 -> h (bf16)
  ln_kernel<<<NTOK, 256, 0, stream>>>(x, ln1_g, ln1_b, h_bf);

  // 3) Q/K/V projections (WMMA, 64x64 per 4-wave block, async B staging)
  dim3 gqkv(DD / 64, NTOK / 64);
  gemm_bf16<DD><<<gqkv, 128, 0, stream>>>(h_bf, wq_bf, bq, nullptr, nullptr, q_bf, nullptr, NTOK, DD);
  gemm_bf16<DD><<<gqkv, 128, 0, stream>>>(h_bf, wk_bf, bk, nullptr, nullptr, k_bf, nullptr, NTOK, DD);
  gemm_bf16<DD><<<gqkv, 128, 0, stream>>>(h_bf, wv_bf, bv, nullptr, nullptr, nullptr, vt_bf, NTOK, DD);

  // 4) attention (WMMA scores + WMMA P@V on transposed V)
  attn_kernel<<<dim3(TT / 16, HH, BB), 32, 0, stream>>>(q_bf, k_bf, vt_bf, ao_bf);

  // 5) output projection + bias + residual -> x1 (fp32)
  gemm_bf16<DD><<<gqkv, 128, 0, stream>>>(ao_bf, wo_bf, bo, x, x1, nullptr, nullptr, NTOK, DD);

  // 6) LN2 -> h2 (bf16), PKM query projection (WMMA)
  ln_kernel<<<NTOK, 256, 0, stream>>>(x1, ln2_g, ln2_b, h2_bf);
  gemm_bf16<DD><<<dim3(DQW / 64, NTOK / 64), 128, 0, stream>>>(h2_bf, pwq_bf, nullptr, nullptr,
                                                               qf, nullptr, nullptr, NTOK, DQW);

  // 7) BatchNorm over tokens
  bn_stats<<<DQW, 256, 0, stream>>>(qf, mu, rs);
  bn_apply<<<(int)(((size_t)NTOK * DQW + 255) / 256), 256, 0, stream>>>(qf, mu, rs, bn_g, bn_b, qn);

  // 8) sub-key dots on the fp32 matrix pipe (v_wmma_f32_16x16x4_f32)
  pkm_dots_wmma<<<dim3(NTOK / 16, NKN / 16, PHN * 2), 32, 0, stream>>>(qn, keys, dots);

  // 9) per-codebook top-32
  pkm_topk<<<dim3(NTOK, PHN, 2), 32, 0, stream>>>(dots, ts, ti);

  // 10) cartesian top-32, softmax, value gather + residual -> out
  pkm_combine<<<NTOK, 256, 0, stream>>>(ts, ti, values, x1, (float*)d_out);
}
